// SwinIR_14224931684598
// MI455X (gfx1250) — matrix operand order repaired and checked by reading, not compiled, not run on hardware
//
#include <hip/hip_runtime.h>
#include <math.h>

typedef __attribute__((ext_vector_type(16))) _Float16 v16h;
typedef __attribute__((ext_vector_type(8)))  _Float16 v8h;
typedef __attribute__((ext_vector_type(8)))  float    v8f;

#define HP    136            // padded H=W
#define NTOK  (HP*HP)        // 18496 tokens
#define CDIM  180
#define NHEAD 6
#define HDIM  30
#define WSZ   8
#define NWIN  17             // HP/WSZ

// padded strides (multiples of 32 for guard-free K loops, 16B-aligned rows)
#define LD_Y   192           // LN output / attn output row stride (180 -> 192)
#define LD_MLP 384           // fc1 output row stride (360 -> 384)

// ---------------------------------------------------------------------------
// WMMA fragment loaders (CDNA5 v_wmma_f32_16x16x32_f16 layouts, ISA 7.12.2)
// A 16x32 per-lane K pattern is contiguous: a[h] = row[8g+h], row[16+8g+h]
// -> two ds_load_b128 per fragment.
// ---------------------------------------------------------------------------
__device__ __forceinline__ v16h frag_A_rowmajor(const _Float16* base, int ld) {
  int l = threadIdx.x & 31;
  int g = l >> 4, m = l & 15;
  const _Float16* row = base + (size_t)m * ld;
  v8h lo = *(const v8h*)(row + 8 * g);        // k = 8g .. 8g+7
  v8h hi = *(const v8h*)(row + 16 + 8 * g);   // k = 16+8g .. 16+8g+7
  v16h a;
#pragma unroll
  for (int i = 0; i < 8; ++i) { a[i] = lo[i]; a[8 + i] = hi[i]; }
  return a;
}

// B fragment from k-major storage: element (k,n) at base[n*ld + k];
// per-lane k = 8g + h, 16 + 8g + h (same as A) -> two ds_load_b128.
__device__ __forceinline__ v16h frag_B_kmajor(const _Float16* base, int ld) {
  int l = threadIdx.x & 31;
  int g = l >> 4, n = l & 15;
  const _Float16* col = base + (size_t)n * ld;
  v8h lo = *(const v8h*)(col + 8 * g);
  v8h hi = *(const v8h*)(col + 16 + 8 * g);
  v16h b;
#pragma unroll
  for (int i = 0; i < 8; ++i) { b[i] = lo[i]; b[8 + i] = hi[i]; }
  return b;
}

__device__ __forceinline__ v8h cvt8(float4 f0, float4 f1) {
  v8h h;
  h[0] = (_Float16)f0.x; h[1] = (_Float16)f0.y; h[2] = (_Float16)f0.z; h[3] = (_Float16)f0.w;
  h[4] = (_Float16)f1.x; h[5] = (_Float16)f1.y; h[6] = (_Float16)f1.z; h[7] = (_Float16)f1.w;
  return h;
}

// ---------------------------------------------------------------------------
// Weight pre-pass: f32 row-major W[K][N] -> zero-padded k-major f16
// out[nPad][kPad] (one-time per launch; removes all guards from GEMM loop)
// ---------------------------------------------------------------------------
__global__ void wcvt_k(const float* __restrict__ W, _Float16* __restrict__ out,
                       int K, int N, int Kpad, int Npad)
{
  int idx = blockIdx.x * blockDim.x + threadIdx.x;
  if (idx >= Npad * Kpad) return;
  int n = idx / Kpad, k = idx % Kpad;
  out[idx] = (k < K && n < N) ? (_Float16)W[(size_t)k * N + n] : (_Float16)0.f;
}

// zero pad columns [n0, ld) of a row-major [M][ld] f32 buffer
__global__ void padcols_k(float* __restrict__ p, int M, int n0, int ld)
{
  int w = ld - n0;
  int idx = blockIdx.x * blockDim.x + threadIdx.x;
  if (idx >= M * w) return;
  int r = idx / w, c = idx % w;
  p[(size_t)r * ld + n0 + c] = 0.f;
}

// ---------------------------------------------------------------------------
// GEMM: C[M,N] = act(A @ B + bias + residual)
// A: f32 [M][lda], zero-padded so cols [K, Kpad) are 0; lda == Kpad.
// Bt: f16 k-major [nPad][Kpad], zero-padded (pre-converted by wcvt_k).
// Guard-free staging: A = 2x global_load_b128 + cvt_pk; B = 1x global_load_b128.
// Epilogue: store when gn < ldc; value 0 for gn >= N (zero-fills C's pad).
// act: 0 = none, 1 = exact GELU. M is a multiple of 64.
// ---------------------------------------------------------------------------
__global__ __launch_bounds__(256) void gemm_wmma_k(
    const float* __restrict__ A, int lda,
    const _Float16* __restrict__ Bt, int ldb /*Kpad*/,
    const float* __restrict__ bias, const float* __restrict__ residual,
    float* __restrict__ C, int ldc, int M, int N, int act)
{
  __shared__ __align__(16) _Float16 As[64 * 32];   // [m][k]
  __shared__ __align__(16) _Float16 Bs[64 * 32];   // [n][k]  (k-major)
  const int m0 = blockIdx.y * 64;
  const int n0 = blockIdx.x * 64;
  const int tid = threadIdx.x;
  const int wave = tid >> 5;
  const int m_t = wave >> 1;       // 0..3
  const int n_half = wave & 1;     // 0..1 (two 16-wide N tiles each)
  // staging coordinates
  const int ar = tid >> 2;               // 0..63 : A row
  const int ac = (tid & 3) * 8;          // 0,8,16,24 : A col block
  const int bn = tid & 63;               // 0..63 : B col
  const int bk = (tid >> 6) * 8;         // 0,8,16,24 : B k block
  const float* asrc = A + (size_t)(m0 + ar) * lda + ac;
  const _Float16* bsrc = Bt + (size_t)(n0 + bn) * ldb + bk;
  v8f acc0 = {}; v8f acc1 = {};
  const int nk = ldb >> 5;
  for (int kc = 0; kc < nk; ++kc) {
    const int k0 = kc * 32;
    // ---- stage A (64x32), unconditional vector loads ----
    float4 f0 = *(const float4*)(asrc + k0);
    float4 f1 = *(const float4*)(asrc + k0 + 4);
    *(v8h*)(&As[ar * 32 + ac]) = cvt8(f0, f1);
    // ---- stage B (k-major), one 16B load of 8 halfs ----
    *(v8h*)(&Bs[bn * 32 + bk]) = *(const v8h*)(bsrc + k0);
    if (kc + 1 < nk) {   // prefetch next A chunk (global_prefetch_b8)
      __builtin_prefetch(asrc + k0 + 32, 0, 0);
    }
    __syncthreads();
    v16h a  = frag_A_rowmajor(&As[m_t * 16 * 32], 32);
    v16h b0 = frag_B_kmajor(&Bs[(n_half * 2 + 0) * 16 * 32], 32);
    v16h b1 = frag_B_kmajor(&Bs[(n_half * 2 + 1) * 16 * 32], 32);
    acc0 = __builtin_amdgcn_wmma_f32_16x16x32_f16(false, a, false, b0, (short)0, acc0, false, false);
    acc1 = __builtin_amdgcn_wmma_f32_16x16x32_f16(false, a, false, b1, (short)0, acc1, false, false);
    __syncthreads();
  }
  const int l = tid & 31, g = l >> 4, nn = l & 15;
  auto epilogue = [&](const v8f& accv, int n_t) {
    int gn = n0 + n_t * 16 + nn;
    if (gn >= ldc) return;              // beyond padded row: nothing to write
    bool inN = (gn < N);
    float bv = (bias && inN) ? bias[gn] : 0.f;
#pragma unroll
    for (int r = 0; r < 8; ++r) {
      int gm = m0 + m_t * 16 + r + 8 * g;
      float v = accv[r] + bv;
      if (residual && inN) v += residual[(size_t)gm * ldc + gn];
      if (act == 1) v = 0.5f * v * (1.f + erff(v * 0.70710678118f));
      if (!inN) v = 0.f;                // zero-fill pad columns of C
      C[(size_t)gm * ldc + gn] = v;
    }
  };
  epilogue(acc0, n_half * 2 + 0);
  epilogue(acc1, n_half * 2 + 1);
}

// ---------------------------------------------------------------------------
// LayerNorm over C=180, one wave32 per token; output row stride LD_Y with
// zeroed pad columns (keeps downstream GEMM staging guard-free).
// ---------------------------------------------------------------------------
__global__ __launch_bounds__(256) void layernorm_k(
    const float* __restrict__ x, const float* __restrict__ gam,
    const float* __restrict__ bet, float* __restrict__ y, int ntok)
{
  int wave = threadIdx.x >> 5;
  int l = threadIdx.x & 31;
  int tok = blockIdx.x * 8 + wave;
  if (tok >= ntok) return;
  const float* xr = x + (size_t)tok * CDIM;
  float s = 0.f, s2 = 0.f;
  for (int c = l; c < CDIM; c += 32) { float v = xr[c]; s += v; s2 += v * v; }
#pragma unroll
  for (int off = 16; off > 0; off >>= 1) {
    s  += __shfl_xor(s, off, 32);
    s2 += __shfl_xor(s2, off, 32);
  }
  float mu  = s / (float)CDIM;
  float var = s2 / (float)CDIM - mu * mu;
  float inv = rsqrtf(var + 1e-5f);
  float* yr = y + (size_t)tok * LD_Y;
  for (int c = l; c < LD_Y; c += 32) {
    int cc = (c < CDIM) ? c : (CDIM - 1);
    float v = (xr[cc] - mu) * inv * gam[cc] + bet[cc];
    yr[c] = (c < CDIM) ? v : 0.f;
  }
}

// ---------------------------------------------------------------------------
// Shifted-window attention, one block(128 thr = 4 waves) per (window, head).
// qkv: [NTOK, 540] laid out (3, NHEAD, HDIM). outp: [NTOK, LD_Y] head-major.
// q,k staged [tok][d]; v staged transposed [d][tok] so all B-fragments are
// k-major (vectorized ds_load_b128).
// ---------------------------------------------------------------------------
__global__ __launch_bounds__(128) void win_attn_k(
    const float* __restrict__ qkv, const float* __restrict__ bias_tab,
    float* __restrict__ outp, int shift)
{
  __shared__ __align__(16) _Float16 qs[64 * 32];   // [tok][d]
  __shared__ __align__(16) _Float16 ks[64 * 32];   // [tok][d] == k-major for q*k^T
  __shared__ __align__(16) _Float16 vst[32 * 64];  // [d][tok] == k-major for P*V
  __shared__ __align__(16) float    Ss[64 * 64];
  __shared__ __align__(16) _Float16 Ps[64 * 64];

  const int win = blockIdx.x;
  const int head = blockIdx.y;
  const int wh = win / NWIN, ww = win % NWIN;
  const int tid = threadIdx.x;

  // stage q/k/v (64 tokens x 32 lanes of head-dim, pad 30->32) as f16
  for (int e = tid; e < 64 * 32; e += 128) {
    int t = e >> 5, d = e & 31;
    int i = t >> 3, j = t & 7;
    int hh = wh * WSZ + i, wp = ww * WSZ + j;
    if (shift) { hh = (hh + 4) % HP; wp = (wp + 4) % HP; }
    const float* bse = qkv + ((size_t)hh * HP + wp) * 540 + head * HDIM;
    int dc = (d < HDIM) ? d : (HDIM - 1);
    float msk = (d < HDIM) ? 1.f : 0.f;
    qs[e] = (_Float16)(bse[dc] * msk);
    ks[e] = (_Float16)(bse[180 + dc] * msk);
    vst[d * 64 + t] = (_Float16)(bse[360 + dc] * msk);
  }
  __syncthreads();

  const int wave = tid >> 5;
  const int l = tid & 31, g = l >> 4, nn = l & 15;
  const int m_t = wave;                 // 16-row strip per wave
  const float scale = 0.18257418583f;   // 30^-0.5

  // S = q k^T  (4 tiles per wave), + scale + rel-pos bias + shift mask
  v16h aq = frag_A_rowmajor(&qs[m_t * 16 * 32], 32);
#pragma unroll
  for (int n_t = 0; n_t < 4; ++n_t) {
    v16h bkf = frag_B_kmajor(&ks[n_t * 16 * 32], 32);
    v8f c = {};
    c = __builtin_amdgcn_wmma_f32_16x16x32_f16(false, aq, false, bkf, (short)0, c, false, false);
#pragma unroll
    for (int r = 0; r < 8; ++r) {
      int mloc = m_t * 16 + r + 8 * g;
      int nloc = n_t * 16 + nn;
      int qi = mloc >> 3, qj = mloc & 7;
      int ki = nloc >> 3, kj = nloc & 7;
      float v = c[r] * scale +
                bias_tab[((qi - ki + 7) * 15 + (qj - kj + 7)) * NHEAD + head];
      if (shift) {
        int mh = wh * WSZ + qi, mw = ww * WSZ + qj;
        int kh = wh * WSZ + ki, kw = ww * WSZ + kj;
        int rq = (mh < HP - WSZ ? 0 : (mh < HP - WSZ / 2 ? 1 : 2)) * 3 +
                 (mw < HP - WSZ ? 0 : (mw < HP - WSZ / 2 ? 1 : 2));
        int rk = (kh < HP - WSZ ? 0 : (kh < HP - WSZ / 2 ? 1 : 2)) * 3 +
                 (kw < HP - WSZ ? 0 : (kw < HP - WSZ / 2 ? 1 : 2));
        if (rq != rk) v -= 100.f;
      }
      Ss[mloc * 64 + nloc] = v;
    }
  }
  __syncthreads();

  // row softmax (64 rows, one thread per row)
  if (tid < 64) {
    float mx = -1e30f;
    for (int c2 = 0; c2 < 64; ++c2) mx = fmaxf(mx, Ss[tid * 64 + c2]);
    float sum = 0.f;
    for (int c2 = 0; c2 < 64; ++c2) {
      float e = __expf(Ss[tid * 64 + c2] - mx);
      Ss[tid * 64 + c2] = e; sum += e;
    }
    float inv = 1.f / sum;
    for (int c2 = 0; c2 < 64; ++c2)
      Ps[tid * 64 + c2] = (_Float16)(Ss[tid * 64 + c2] * inv);
  }
  __syncthreads();

  // O = P @ V  (K=64 in two chunks, 2 N-tiles of 16 covering d=0..31)
#pragma unroll
  for (int n_t = 0; n_t < 2; ++n_t) {
    v8f c = {};
#pragma unroll
    for (int kk = 0; kk < 2; ++kk) {
      v16h ap = frag_A_rowmajor(&Ps[m_t * 16 * 64 + kk * 32], 64);
      v16h bv = frag_B_kmajor(&vst[n_t * 16 * 64 + kk * 32], 64);
      c = __builtin_amdgcn_wmma_f32_16x16x32_f16(false, ap, false, bv, (short)0, c, false, false);
    }
#pragma unroll
    for (int r = 0; r < 8; ++r) {
      int mloc = m_t * 16 + r + 8 * g;
      int col = n_t * 16 + nn;
      if (col < HDIM) {
        int i = mloc >> 3, j = mloc & 7;
        int hh = wh * WSZ + i, wp = ww * WSZ + j;
        if (shift) { hh = (hh + 4) % HP; wp = (wp + 4) % HP; }
        outp[((size_t)hh * HP + wp) * LD_Y + head * HDIM + col] = c[r];
      }
    }
  }
}

// ---------------------------------------------------------------------------
// Direct NCHW 3x3 conv (zero pad), optional residual, act: 0 none, 2 LeakyReLU
// ---------------------------------------------------------------------------
__global__ void conv3x3_k(const float* __restrict__ in, const float* __restrict__ w,
                          const float* __restrict__ bias, const float* __restrict__ res,
                          float* __restrict__ out, int Cin, int Cout, int H, int W, int act)
{
  int idx = blockIdx.x * blockDim.x + threadIdx.x;
  int total = Cout * H * W;
  if (idx >= total) return;
  int x = idx % W; int t = idx / W; int y = t % H; int co = t / H;
  float acc = bias[co];
  const float* wr = w + (size_t)co * Cin * 9;
  for (int ci = 0; ci < Cin; ++ci) {
    const float* ip = in + (size_t)ci * H * W;
    const float* wp = wr + ci * 9;
#pragma unroll
    for (int ky = 0; ky < 3; ++ky) {
      int yy = y + ky - 1;
      if (yy < 0 || yy >= H) continue;
#pragma unroll
      for (int kx = 0; kx < 3; ++kx) {
        int xx = x + kx - 1;
        if (xx < 0 || xx >= W) continue;
        acc += wp[ky * 3 + kx] * ip[(size_t)yy * W + xx];
      }
    }
  }
  if (res) acc += res[idx];
  if (act == 2) acc = (acc >= 0.f) ? acc : 0.01f * acc;
  out[idx] = acc;
}

// ---------------------------------------------------------------------------
// Small elementwise kernels
// ---------------------------------------------------------------------------
__global__ void pad_reflect_k(const float* __restrict__ in, float* __restrict__ out) {
  int idx = blockIdx.x * blockDim.x + threadIdx.x;
  if (idx >= 3 * HP * HP) return;
  int x = idx % HP; int t = idx / HP; int y = t % HP; int c = t / HP;
  int sy = (y < 128) ? y : 254 - y;
  int sx = (x < 128) ? x : 254 - x;
  out[idx] = in[((size_t)c * 128 + sy) * 128 + sx];
}

__global__ void nchw2nhwc_k(const float* __restrict__ in, float* __restrict__ out) {
  int idx = blockIdx.x * blockDim.x + threadIdx.x;
  if (idx >= CDIM * NTOK) return;
  int c = idx % CDIM; int p = idx / CDIM;
  out[idx] = in[(size_t)c * NTOK + p];
}

__global__ void nhwc2nchw_k(const float* __restrict__ in, float* __restrict__ out) {
  int idx = blockIdx.x * blockDim.x + threadIdx.x;
  if (idx >= CDIM * NTOK) return;
  int p = idx % NTOK; int c = idx / NTOK;
  out[idx] = in[(size_t)p * CDIM + c];
}

__global__ void pshuffle_k(const float* __restrict__ in, float* __restrict__ out,
                           int Cout, int Hin, int Win) {
  int Ho = Hin * 2, Wo = Win * 2;
  int idx = blockIdx.x * blockDim.x + threadIdx.x;
  int total = Cout * Ho * Wo;
  if (idx >= total) return;
  int x = idx % Wo; int t = idx / Wo; int y = t % Ho; int c = t / Ho;
  int ic = c * 4 + (y & 1) * 2 + (x & 1);
  out[idx] = in[((size_t)ic * Hin + (y >> 1)) * Win + (x >> 1)];
}

__global__ void crop_k(const float* __restrict__ in, float* __restrict__ out) {
  int idx = blockIdx.x * blockDim.x + threadIdx.x;
  if (idx >= 3 * 512 * 512) return;
  int x = idx % 512; int t = idx / 512; int y = t % 512; int c = t / 512;
  out[idx] = in[((size_t)c * 544 + y) * 544 + x];
}

// ---------------------------------------------------------------------------
// Host orchestration
// ---------------------------------------------------------------------------
static inline int cdiv_i(long a, long b) { return (int)((a + b - 1) / b); }

extern "C" void kernel_launch(void* const* d_in, const int* in_sizes, int n_in,
                              void* d_out, int out_size, void* d_ws, size_t ws_size,
                              hipStream_t stream) {
  (void)in_sizes; (void)out_size; (void)ws_size;
  // ---- parameter pointers (jax pytree: dict keys sorted alphabetically) ----
  const float* img    = (const float*)d_in[0];
  const float* deep_b = (const float*)d_in[1];
  const float* deep_w = (const float*)d_in[2];
  const float* rec_b[4]; const float* rec_w[4];
  for (int i = 0; i < 4; ++i) {
    rec_b[i] = (const float*)d_in[3 + 2 * i];
    rec_w[i] = (const float*)d_in[4 + 2 * i];
  }
  const float* shallow_b = (const float*)d_in[n_in - 2];
  const float* shallow_w = (const float*)d_in[n_in - 1];

  // ---- workspace layout (floats); ~274 MB, recon aliases the pool ----
  const size_t SZF = (size_t)CDIM * NTOK;            // 3,329,280
  float* W0   = (float*)d_ws;
  float* hA   = W0;                                   // NCHW feature (persists)
  float* padb = hA + SZF;                             // pool base
  float* lf   = padb + (size_t)3 * HP * HP;
  float* xb   = lf + SZF;                             // NHWC tokens [NTOK][180]
  float* yb   = xb + SZF;                             // LN out     [NTOK][192]
  float* qkvb = yb + (size_t)NTOK * LD_Y;             // [NTOK][540]
  float* attnb= qkvb + (size_t)NTOK * 540;            // [NTOK][192]
  float* zb   = attnb + (size_t)NTOK * LD_Y;          // [NTOK][180]
  float* mlpb = zb + SZF;                             // [NTOK][384]
  float* ctmp = mlpb + (size_t)NTOK * LD_MLP;         // NCHW temp
  float* hB   = ctmp + SZF;
  _Float16* warena = (_Float16*)(hB + SZF);           // f16 k-major weights
  // per-STL arena: qkv 576x192, proj 192x192, fc1 384x192, fc2 192x384
  const size_t WQ = 576 * 192, WP = 192 * 192, W1 = 384 * 192, W2 = 192 * 384;
  const size_t WSTL = WQ + WP + W1 + W2;              // 294,912 halfs
  float* r4 = hB + SZF + (WSTL * 36 + 1) / 2;         // 64*544*544
  // recon aliases (transformer pool is dead by then; r3 ends below ctmp)
  float* r0 = padb;                                   // 64*136*136
  float* r1 = r0 + (size_t)64 * 136 * 136;            // 256*136*136
  float* r2 = r1 + (size_t)256 * 136 * 136;           // 64*272*272
  float* r3 = r2 + (size_t)64 * 272 * 272;            // 256*272*272
  float* r5 = padb;                                   // 3*544*544 (r0 dead)

  const int B256 = 256;

  // ---- weight pre-pass: convert all STL weights to padded k-major f16 ----
  for (int r = 0; r < 6; ++r) {
    for (int s = 0; s < 6; ++s) {
      int sb = 11 + 74 * r + 2 + 12 * s;
      _Float16* wa = warena + (size_t)(r * 6 + s) * WSTL;
      const float* fc1_w  = (const float*)d_in[sb + 2];
      const float* fc2_w  = (const float*)d_in[sb + 4];
      const float* proj_w = (const float*)d_in[sb + 10];
      const float* qkv_w  = (const float*)d_in[sb + 11];
      wcvt_k<<<cdiv_i(WQ, B256), B256, 0, stream>>>(qkv_w,  wa,                180, 540, 192, 576);
      wcvt_k<<<cdiv_i(WP, B256), B256, 0, stream>>>(proj_w, wa + WQ,           180, 180, 192, 192);
      wcvt_k<<<cdiv_i(W1, B256), B256, 0, stream>>>(fc1_w,  wa + WQ + WP,      180, 360, 192, 384);
      wcvt_k<<<cdiv_i(W2, B256), B256, 0, stream>>>(fc2_w,  wa + WQ + WP + W1, 360, 180, 384, 192);
    }
  }

  // ---- shallow stage ----
  pad_reflect_k<<<cdiv_i(3 * HP * HP, B256), B256, 0, stream>>>(img, padb);
  conv3x3_k<<<cdiv_i((long)CDIM * HP * HP, B256), B256, 0, stream>>>(
      padb, shallow_w, shallow_b, nullptr, lf, 3, CDIM, HP, HP, 0);
  hipMemcpyAsync(hA, lf, SZF * sizeof(float), hipMemcpyDeviceToDevice, stream);

  // ---- 6 RSTBs x 6 STL blocks ----
  const int TBLK = cdiv_i(NTOK, 8);
  const dim3 gQKV(cdiv_i(540, 64), NTOK / 64);
  const dim3 gPRJ(cdiv_i(180, 64), NTOK / 64);
  const dim3 gFC1(cdiv_i(360, 64), NTOK / 64);
  const dim3 gATT(NWIN * NWIN, NHEAD);
  const int PADBLK = cdiv_i((long)NTOK * (LD_Y - CDIM), B256);

  for (int r = 0; r < 6; ++r) {
    int rb = 11 + 74 * r;
    const float* conv_b = (const float*)d_in[rb + 0];
    const float* conv_w = (const float*)d_in[rb + 1];
    float* src = (r & 1) ? hB : hA;
    float* dst = (r & 1) ? hA : hB;

    nchw2nhwc_k<<<cdiv_i(SZF, B256), B256, 0, stream>>>(src, xb);

    for (int s = 0; s < 6; ++s) {
      int sb = rb + 2 + 12 * s;
      const float* bias_tab = (const float*)d_in[sb + 0];
      const float* fc1_b  = (const float*)d_in[sb + 1];
      const float* fc2_b  = (const float*)d_in[sb + 3];
      const float* ln1_b  = (const float*)d_in[sb + 5];
      const float* ln1_g  = (const float*)d_in[sb + 6];
      const float* ln2_b  = (const float*)d_in[sb + 7];
      const float* ln2_g  = (const float*)d_in[sb + 8];
      const float* proj_b = (const float*)d_in[sb + 9];
      _Float16* wa = warena + (size_t)(r * 6 + s) * WSTL;
      const _Float16* wq = wa;
      const _Float16* wp = wa + WQ;
      const _Float16* w1 = wa + WQ + WP;
      const _Float16* w2 = wa + WQ + WP + W1;
      int shift = (s % 2 != 0) ? 1 : 0;

      layernorm_k<<<TBLK, B256, 0, stream>>>(xb, ln1_g, ln1_b, yb, NTOK);
      gemm_wmma_k<<<gQKV, B256, 0, stream>>>(yb, LD_Y, wq, 192, nullptr, nullptr,
                                             qkvb, 540, NTOK, 540, 0);
      win_attn_k<<<gATT, 128, 0, stream>>>(qkvb, bias_tab, attnb, shift);
      padcols_k<<<PADBLK, B256, 0, stream>>>(attnb, NTOK, CDIM, LD_Y);
      gemm_wmma_k<<<gPRJ, B256, 0, stream>>>(attnb, LD_Y, wp, 192, proj_b, xb,
                                             zb, CDIM, NTOK, CDIM, 0);
      layernorm_k<<<TBLK, B256, 0, stream>>>(zb, ln2_g, ln2_b, yb, NTOK);
      gemm_wmma_k<<<gFC1, B256, 0, stream>>>(yb, LD_Y, w1, 192, fc1_b, nullptr,
                                             mlpb, LD_MLP, NTOK, 360, 1 /*gelu*/);
      gemm_wmma_k<<<gPRJ, B256, 0, stream>>>(mlpb, LD_MLP, w2, 384, fc2_b, zb,
                                             xb, CDIM, NTOK, CDIM, 0);
    }

    nhwc2nchw_k<<<cdiv_i(SZF, B256), B256, 0, stream>>>(xb, ctmp);
    conv3x3_k<<<cdiv_i((long)CDIM * HP * HP, B256), B256, 0, stream>>>(
        ctmp, conv_w, conv_b, src, dst, CDIM, CDIM, HP, HP, 0);
  }
  // after 6 RSTBs final h is in hA

  // ---- deep conv + global residual: z0 = conv(h) + lf -> ctmp ----
  conv3x3_k<<<cdiv_i((long)CDIM * HP * HP, B256), B256, 0, stream>>>(
      hA, deep_w, deep_b, lf, ctmp, CDIM, CDIM, HP, HP, 0);

  // ---- reconstruction head ----
  conv3x3_k<<<cdiv_i((long)64 * HP * HP, B256), B256, 0, stream>>>(
      ctmp, rec_w[0], rec_b[0], nullptr, r0, CDIM, 64, HP, HP, 2 /*leaky*/);
  conv3x3_k<<<cdiv_i((long)256 * HP * HP, B256), B256, 0, stream>>>(
      r0, rec_w[1], rec_b[1], nullptr, r1, 64, 256, HP, HP, 0);
  pshuffle_k<<<cdiv_i((long)64 * 272 * 272, B256), B256, 0, stream>>>(
      r1, r2, 64, HP, HP);
  conv3x3_k<<<cdiv_i((long)256 * 272 * 272, B256), B256, 0, stream>>>(
      r2, rec_w[2], rec_b[2], nullptr, r3, 64, 256, 272, 272, 0);
  pshuffle_k<<<cdiv_i((long)64 * 544 * 544, B256), B256, 0, stream>>>(
      r3, r4, 64, 272, 272);
  conv3x3_k<<<cdiv_i((long)3 * 544 * 544, B256), B256, 0, stream>>>(
      r4, rec_w[3], rec_b[3], nullptr, r5, 64, 3, 544, 544, 0);
  crop_k<<<cdiv_i(3 * 512 * 512, B256), B256, 0, stream>>>(r5, (float*)d_out);
}
